// Quantize_24764781429538
// MI455X (gfx1250) — matrix operand order repaired and checked by reading, not compiled, not run on hardware
//
#include <hip/hip_runtime.h>
#include <hip/hip_bf16.h>

// ---------------------------------------------------------------------------
// Fused  project -> cosine-similarity -> per-codebook argmax  for MI455X
// (gfx1250, wave32, v_wmma_f32_16x16x32_bf16, f32 accumulate).
//
//   argmax_m (x.cb_m)/(|x||cb_m|) == argmax_m (x.cbn_m)   -> x never normalized
//
// v2: codebook B-tiles are staged ONCE per block into LDS with double-buffered
// async copies (global_load_async_to_lds_b128 / s_wait_asynccnt) and shared by
// all 8 waves -> GEMM2 L2 traffic drops from ~2 GB to ~256 MB. GEMM1 A-frags
// are built straight from global f32 (no LDS staging pass needed).
// ---------------------------------------------------------------------------

typedef __bf16 bf16_t;
typedef __attribute__((ext_vector_type(16))) __bf16 v16bf;
typedef __attribute__((ext_vector_type(8)))  __bf16 v8bf;
typedef __attribute__((ext_vector_type(8)))  float   v8f;

#define K_IN   256   // input_dim
#define D      128   // embed_dim
#define M_EMB  512   // entries per codebook
#define N_CB   4     // codebooks
#define ROWS   128   // rows per block (16 per wave)
#define NT     256   // threads per block = 8 waves
#define STEP_E 64    // codebook entries staged per pipeline step
#define STEP_BYTES (STEP_E * D * 2)            // 16 KB
#define NSTEPS ((N_CB * M_EMB) / STEP_E)       // 32

union BF16x16 { v16bf v; v8bf h[2]; bf16_t e[16]; };

// ---- gfx1250 async global->LDS copy (tracked by ASYNCcnt) ------------------
static __device__ __forceinline__ void async_copy_b128(unsigned lds_off, const void* gptr) {
    asm volatile("global_load_async_to_lds_b128 %0, %1, off"
                 :: "v"(lds_off), "v"(gptr) : "memory");
}
static __device__ __forceinline__ void wait_asynccnt0() {
    asm volatile("s_wait_asynccnt 0" ::: "memory");
}
// Generic->LDS offset: LDS aperture maps addr[31:0] to the LDS byte address.
static __device__ __forceinline__ unsigned lds_off_of(const void* p) {
    return (unsigned)(unsigned long long)p;
}

// Load a 16x32 bf16 A-fragment from a row-major bf16 row in LDS.
// Layout: lane holds row (lane&15); elems 0..15 at K=k0+16*hi+e
// -> two contiguous 16-byte chunks.
static __device__ __forceinline__ v16bf load_afrag_lds(const bf16_t* rowp, int k0, int hi) {
    BF16x16 u;
    u.h[0] = *(const v8bf*)(rowp + k0 + 16 * hi);
    u.h[1] = *(const v8bf*)(rowp + k0 + 16 * hi + 8);
    return u.v;
}

// --------------------------- prep kernels ----------------------------------

// Normalize each of the 2048 codebook rows (len 128), emit bf16. 1 wave/row.
__global__ __launch_bounds__(128) void prep_codebook(const float* __restrict__ cb,
                                                     bf16_t* __restrict__ cbn) {
    const int row  = blockIdx.x * 4 + (threadIdx.x >> 5);
    const int lane = threadIdx.x & 31;
    const float* src = cb + (long)row * D;
    float v[4], s = 0.f;
#pragma unroll
    for (int i = 0; i < 4; ++i) { v[i] = src[lane + 32 * i]; s += v[i] * v[i]; }
#pragma unroll
    for (int off = 16; off >= 1; off >>= 1) s += __shfl_xor(s, off, 32);
    const float inv = rsqrtf(s);
    bf16_t* dst = cbn + (long)row * D;
#pragma unroll
    for (int i = 0; i < 4; ++i) dst[lane + 32 * i] = (bf16_t)(v[i] * inv);
}

// Transpose projector [256][128] f32 -> projT [128][256] bf16 (contiguous
// 32-byte B-fragment loads per lane).
__global__ __launch_bounds__(256) void prep_projT(const float* __restrict__ proj,
                                                  bf16_t* __restrict__ projT) {
    const int idx = blockIdx.x * 256 + threadIdx.x;   // 0..32767
    const int k = idx >> 7;                           // 0..255
    const int n = idx & (D - 1);                      // 0..127
    projT[n * K_IN + k] = (bf16_t)proj[idx];
}

// --------------------------- fused main kernel -----------------------------

__global__ __launch_bounds__(NT) void fused_vq(const float* __restrict__ in,
                                               const bf16_t* __restrict__ projT, // [128][256]
                                               const bf16_t* __restrict__ cbn,   // [2048][128]
                                               int* __restrict__ out) {          // [65536][4]
    __shared__ __align__(32) bf16_t s_x[ROWS * D];          // 32 KB projected tile
    __shared__ __align__(32) bf16_t s_b[2][STEP_E * D];     // 2 x 16 KB codebook buffers

    const int  tid   = threadIdx.x;
    const int  wave  = tid >> 5;
    const int  lane  = tid & 31;
    const int  lo    = lane & 15;
    const int  hi    = lane >> 4;
    const long row0  = (long)blockIdx.x * ROWS;
    const int  wrow0 = wave * 16;

    // ---- GEMM1: x[16][128] = in_rows[16][256] @ projector -----------------
    // A-fragments straight from global f32 (each input float read exactly once).
    {
        v16bf a[8];
        const float* arow = in + (row0 + wrow0 + lo) * (long)K_IN;
#pragma unroll
        for (int kc = 0; kc < 8; ++kc) {
            const int b0 = kc * 32 + 16 * hi;
            const float4 f0 = *(const float4*)(arow + b0);
            const float4 f1 = *(const float4*)(arow + b0 + 4);
            const float4 f2 = *(const float4*)(arow + b0 + 8);
            const float4 f3 = *(const float4*)(arow + b0 + 12);
            BF16x16 u;
            u.e[0]  = (bf16_t)f0.x; u.e[1]  = (bf16_t)f0.y;
            u.e[2]  = (bf16_t)f0.z; u.e[3]  = (bf16_t)f0.w;
            u.e[4]  = (bf16_t)f1.x; u.e[5]  = (bf16_t)f1.y;
            u.e[6]  = (bf16_t)f1.z; u.e[7]  = (bf16_t)f1.w;
            u.e[8]  = (bf16_t)f2.x; u.e[9]  = (bf16_t)f2.y;
            u.e[10] = (bf16_t)f2.z; u.e[11] = (bf16_t)f2.w;
            u.e[12] = (bf16_t)f3.x; u.e[13] = (bf16_t)f3.y;
            u.e[14] = (bf16_t)f3.z; u.e[15] = (bf16_t)f3.w;
            a[kc] = u.v;
        }

#pragma unroll
        for (int j = 0; j < 8; ++j) {                 // 8 N-subtiles of 16
            v8f acc = {};
            const bf16_t* bp = projT + (j * 16 + lo) * K_IN + 16 * hi;
#pragma unroll
            for (int kc = 0; kc < 8; ++kc) {
                const v16bf b = *(const v16bf*)(bp + kc * 32);
                acc = __builtin_amdgcn_wmma_f32_16x16x32_bf16(
                        false, a[kc], false, b, (short)0, acc, false, false);
            }
            // C layout: lane holds col j*16+lo, rows g+8*hi -> scatter to LDS
#pragma unroll
            for (int g = 0; g < 8; ++g) {
                const int r = wrow0 + g + 8 * hi;
                s_x[r * D + j * 16 + lo] = (bf16_t)acc[g];
            }
        }
    }

    // Kick off async staging of the first codebook chunk (doesn't touch s_x).
    {
        const char* g = (const char*)cbn;
        char* l = (char*)&s_b[0][0];
#pragma unroll
        for (int i = 0; i < STEP_BYTES / (NT * 16); ++i) {     // 4 x 16B per thread
            const unsigned off = (unsigned)(tid * 16 + i * NT * 16);
            async_copy_b128(lds_off_of(l + off), g + off);
        }
    }
    __syncthreads();                                  // s_x visible to this wave's frags

    // ---- GEMM2 + per-codebook argmax --------------------------------------
    v16bf xf[4];
    {
        const bf16_t* xrow = s_x + (wrow0 + lo) * D;
#pragma unroll
        for (int kc = 0; kc < 4; ++kc) xf[kc] = load_afrag_lds(xrow, kc * 32, hi);
    }

    int buf = 0;
#pragma unroll 1
    for (int c = 0; c < N_CB; ++c) {
        float bv[8]; int bi[8];
#pragma unroll
        for (int g = 0; g < 8; ++g) { bv[g] = -3.402823466e38f; bi[g] = 0; }

#pragma unroll 1
        for (int st = 0; st < M_EMB / STEP_E; ++st) {          // 8 steps of 64 entries
            const int s = c * (M_EMB / STEP_E) + st;           // global step 0..31
            wait_asynccnt0();                                  // this wave's stage done
            __syncthreads();                                   // all waves' stages done,
                                                               // prev buffer reads retired
            if (s + 1 < NSTEPS) {                              // stage next chunk
                const char* g = (const char*)cbn + (size_t)(s + 1) * STEP_BYTES;
                char* l = (char*)&s_b[buf ^ 1][0];
#pragma unroll
                for (int i = 0; i < STEP_BYTES / (NT * 16); ++i) {
                    const unsigned off = (unsigned)(tid * 16 + i * NT * 16);
                    async_copy_b128(lds_off_of(l + off), g + off);
                }
            }

            const bf16_t* sb = &s_b[buf][0];
#pragma unroll
            for (int nt = 0; nt < STEP_E / 16; ++nt) {         // 4 tiles of 16 entries
                const bf16_t* bp = sb + (nt * 16 + lo) * D + 16 * hi;
                v8f acc = {};
#pragma unroll
                for (int kc = 0; kc < 4; ++kc) {
                    const v16bf b = *(const v16bf*)(bp + kc * 32);
                    acc = __builtin_amdgcn_wmma_f32_16x16x32_bf16(
                            false, xf[kc], false, b, (short)0, acc, false, false);
                }
                const int midx = st * STEP_E + nt * 16 + lo;   // entry idx in codebook
#pragma unroll
                for (int g = 0; g < 8; ++g) {
                    const float v = acc[g];
                    if (v > bv[g]) { bv[g] = v; bi[g] = midx; }
                }
            }
            buf ^= 1;
        }

        // butterfly reduce across the 16 m-columns (stays inside 16-lane halves)
#pragma unroll
        for (int g = 0; g < 8; ++g) {
            float v = bv[g]; int i = bi[g];
#pragma unroll
            for (int off = 8; off >= 1; off >>= 1) {
                const float vv = __shfl_xor(v, off, 16);
                const int   ii = __shfl_xor(i, off, 16);
                if (vv > v || (vv == v && ii < i)) { v = vv; i = ii; }
            }
            if (lo == 0) {
                const long row = row0 + wrow0 + g + 8 * hi;
                out[row * N_CB + c] = i;
            }
        }
    }
}

// --------------------------- launch ----------------------------------------

extern "C" void kernel_launch(void* const* d_in, const int* in_sizes, int n_in,
                              void* d_out, int out_size, void* d_ws, size_t ws_size,
                              hipStream_t stream) {
    (void)in_sizes; (void)n_in; (void)out_size; (void)ws_size;

    const float* in   = (const float*)d_in[0];   // [8,16,512,256] = [65536][256]
    const float* proj = (const float*)d_in[1];   // [256][128]
    const float* cb   = (const float*)d_in[2];   // [4][512][128] = [2048][128]
    int*         out  = (int*)d_out;             // [65536][4]

    bf16_t* cbn   = (bf16_t*)d_ws;                                         // 512 KB
    bf16_t* projT = (bf16_t*)((char*)d_ws + (size_t)N_CB * M_EMB * D * 2); // 64 KB

    prep_codebook<<<(N_CB * M_EMB) / 4, 128, 0, stream>>>(cb, cbn);
    prep_projT<<<(K_IN * D) / 256, 256, 0, stream>>>(proj, projT);

    const int nrows = 8 * 16 * 512;               // 65536
    fused_vq<<<nrows / ROWS, NT, 0, stream>>>(in, projT, cbn, out);
}
